// SamplerAttnFCN_48155173323056
// MI455X (gfx1250) — matrix operand order repaired and checked by reading, not compiled, not run on hardware
//
#include <hip/hip_runtime.h>
#include <hip/hip_bf16.h>
#include <math.h>

// ---------------------------------------------------------------------------
// GAT (3 layers) + FC head for a single query node, CDNA5 (gfx1250, wave32).
// - Heavy GEMMs (x@W1, h1@W2) on v_wmma_f32_16x16x32_bf16 (f32 accumulate).
// - Adjacency rows staged into LDS by the Tensor Data Mover
//   (tensor_load_to_lds + s_wait_tensorcnt), read twice from LDS.
// - Layer 3 algebraically collapsed to a weighted row-average of h2.
// ---------------------------------------------------------------------------

#define NN    2048
#define IND   2049
#define KP1   2080   // 2049 padded to multiple of 32
#define HIDN  32

typedef __attribute__((ext_vector_type(8)))  float        v8f;
typedef __attribute__((ext_vector_type(8)))  __bf16       v8bf;
typedef __attribute__((ext_vector_type(16))) __bf16       v16bf;
typedef __attribute__((ext_vector_type(4)))  unsigned int ui32x4;
typedef __attribute__((ext_vector_type(8)))  int          i32x8;
typedef __attribute__((ext_vector_type(4)))  int          i32x4;

__device__ __forceinline__ unsigned short f2bf(float f) {
    unsigned int u = __float_as_uint(f);
    unsigned int r = (u + 0x7fffu + ((u >> 16) & 1u)) >> 16;  // RNE
    return (unsigned short)r;
}

// ---- TDM: DMA one contiguous row of `nbytes` bytes from global into LDS ----
// D# per CDNA5 ISA (2-D tensor: 1 row x nbytes, 1B elems).
// Caller must be wave-uniform; follow with s_wait_tensorcnt(0) + barrier.
// Toolchain here is the 6-arg clang-23 form:
//   (uint32x4 g0, int32x8 g1, int32x4, int32x4, int32x8, i32 cpol)
__device__ __forceinline__ void tdm_row_to_lds(const unsigned char* gsrc,
                                               void* lds_dst, int nbytes) {
    unsigned long long ga = (unsigned long long)(size_t)gsrc;
    ui32x4 g0;
    g0[0] = 1u;                                           // count=1, user mode
    g0[1] = (unsigned)(size_t)lds_dst;                    // lds_addr (low 32 of flat = LDS offset)
    g0[2] = (unsigned)(ga & 0xffffffffull);               // global_addr[31:0]
    g0[3] = (unsigned)((ga >> 32) & 0x1ffffffull)         // global_addr[56:32]
          | (2u << 30);                                   // type = 2 ("image")
    i32x8 g1;
    g1[0] = 0;                                            // wg_mask=0, data_size=1B, no flags
    g1[1] = (int)((unsigned)nbytes << 16);                // tensor_dim0[15:0] @ bits63:48
    g1[2] = (int)(1u << 16);                              // tensor_dim1=1 @ bits95:80
    g1[3] = (int)((unsigned)nbytes << 16);                // tile_dim0 @ bits127:112
    g1[4] = 1;                                            // tile_dim1=1 (tile_dim2=0)
    g1[5] = nbytes;                                       // tensor_dim0_stride[31:0]
    g1[6] = 0;                                            // stride hi / dim1_stride lo
    g1[7] = 0;
    i32x4 gz4 = {0, 0, 0, 0};
    i32x8 gz8 = {0, 0, 0, 0, 0, 0, 0, 0};
    __builtin_amdgcn_tensor_load_to_lds(g0, g1, gz4, gz4, gz8, 0);
}

// ---- pack x (f32 [NN][IND]) -> bf16 [NN][KP1], zero-padded K ---------------
__global__ void pack_x_kernel(const float* __restrict__ x, unsigned short* __restrict__ xb) {
    int idx = blockIdx.x * 256 + threadIdx.x;
    if (idx >= NN * KP1) return;
    int i = idx / KP1, k = idx % KP1;
    float v = (k < IND) ? x[(size_t)i * IND + k] : 0.f;
    xb[idx] = f2bf(v);
}

// ---- pack W (f32 [Ksrc][Ncols] row-major) -> bf16 transposed [Ncols][Kpad] -
__global__ void pack_wT_kernel(const float* __restrict__ w, unsigned short* __restrict__ wt,
                               int Ksrc, int Kpad, int Ncols) {
    int idx = blockIdx.x * 256 + threadIdx.x;
    if (idx >= Ncols * Kpad) return;
    int n = idx / Kpad, k = idx % Kpad;
    float v = (k < Ksrc) ? w[(size_t)k * Ncols + n] : 0.f;
    wt[idx] = f2bf(v);
}

// ---- C[M][Ncols](f32) = A[M][Kpad](bf16) @ Bt[Ncols][Kpad](bf16, transposed)
// one wave (32 lanes) per 16x16 output tile; K stepped 32 per WMMA.
__global__ __launch_bounds__(32) void gemm_bf16_wmma(const unsigned short* __restrict__ A,
                                                     const unsigned short* __restrict__ Bt,
                                                     float* __restrict__ C,
                                                     int Kpad, int Ncols) {
    const int lane = threadIdx.x;
    const int half = lane >> 4;   // K-half selector for this lane group
    const int lid  = lane & 15;
    const int m0 = blockIdx.x * 16, n0 = blockIdx.y * 16;

    const unsigned short* arow = A  + (size_t)(m0 + lid) * Kpad;
    const unsigned short* brow = Bt + (size_t)(n0 + lid) * Kpad;

    v8f acc = (v8f)0.0f;
    for (int kb = 0; kb < Kpad; kb += 32) {
        // A 16x32 bf16 tile: lane(m=lid), K = half*8..+7 and 16+half*8..+7
        v8bf alo = *(const v8bf*)(arow + kb + half * 8);
        v8bf ahi = *(const v8bf*)(arow + kb + 16 + half * 8);
        v16bf a = __builtin_shufflevector(alo, ahi,
                    0,1,2,3,4,5,6,7,8,9,10,11,12,13,14,15);
        // B 32x16 bf16 tile: lane(n=lid), K = half*8..+7 and 16+half*8..+7
        v8bf blo = *(const v8bf*)(brow + kb + half * 8);
        v8bf bhi = *(const v8bf*)(brow + kb + 16 + half * 8);
        v16bf b = __builtin_shufflevector(blo, bhi,
                    0,1,2,3,4,5,6,7,8,9,10,11,12,13,14,15);
        acc = __builtin_amdgcn_wmma_f32_16x16x32_bf16(false, a, false, b,
                                                      (short)0, acc, false, false);
    }
    // D: VGPR r -> row (m0 + r + 8*half), col (n0 + lid)
    float* crow = C + (size_t)(m0 + half * 8) * Ncols + n0 + lid;
#pragma unroll
    for (int r = 0; r < 8; ++r) crow[(size_t)r * Ncols] = acc[r];
}

// ---- per-(node,head) attention logit halves: gl/gr = g[i,h,:] . a ----------
__global__ void glr_kernel(const float* __restrict__ g, const float* __restrict__ al,
                           const float* __restrict__ ar, float* __restrict__ gl,
                           float* __restrict__ gr) {
    int idx = blockIdx.x * 256 + threadIdx.x;          // node*8 + head
    if (idx >= NN * 8) return;
    const float* gp = g + ((size_t)(idx >> 3)) * HIDN + (idx & 7) * 4;
    float sl = 0.f, sr = 0.f;
#pragma unroll
    for (int d = 0; d < 4; ++d) { sl += gp[d] * al[d]; sr += gp[d] * ar[d]; }
    gl[idx] = sl; gr[idx] = sr;
}

// ---- GAT attention layer (8 heads, d=4): one block per node i --------------
__global__ __launch_bounds__(256) void gat_attn_kernel(const float* __restrict__ g,
                                                       const float* __restrict__ gl,
                                                       const float* __restrict__ gr,
                                                       const unsigned char* __restrict__ adj,
                                                       float* __restrict__ hf,          // optional f32 out
                                                       unsigned short* __restrict__ hb) // optional bf16 out
{
    __shared__ float buf[256 * 40];
    __shared__ unsigned char amask[NN];
    const int i = blockIdx.x, tid = threadIdx.x;
    __builtin_prefetch(gr + (size_t)tid * 8, 0, 1);

    // stage adjacency row into LDS via Tensor Data Mover (wave 0 only; TDM
    // ignores EXEC, gate with a wave-uniform branch, then publish via barrier)
    if (tid < 32) {
        tdm_row_to_lds(adj + (size_t)i * NN, amask, NN);
        __builtin_amdgcn_s_wait_tensorcnt((short)0);
    }
    __syncthreads();

    float gli[8];
#pragma unroll
    for (int h = 0; h < 8; ++h) gli[h] = gl[(size_t)i * 8 + h];

    // pass 1: per-head masked max of leaky-relu logits
    float m[8];
#pragma unroll
    for (int h = 0; h < 8; ++h) m[h] = -INFINITY;
    for (int j = tid; j < NN; j += 256) {
        if (!amask[j]) continue;
        const float* grj = gr + (size_t)j * 8;
#pragma unroll
        for (int h = 0; h < 8; ++h) {
            float s = gli[h] + grj[h];
            s = (s >= 0.f) ? s : 0.2f * s;
            m[h] = fmaxf(m[h], s);
        }
    }
#pragma unroll
    for (int h = 0; h < 8; ++h) buf[tid * 8 + h] = m[h];
    __syncthreads();
    for (int s = 128; s > 0; s >>= 1) {
        if (tid < s)
#pragma unroll
            for (int h = 0; h < 8; ++h)
                buf[tid * 8 + h] = fmaxf(buf[tid * 8 + h], buf[(tid + s) * 8 + h]);
        __syncthreads();
    }
    float mh[8];
#pragma unroll
    for (int h = 0; h < 8; ++h) mh[h] = buf[h];
    __syncthreads();

    // pass 2: unnormalized weighted aggregation + partition function
    float z[8];  float acc[32];
#pragma unroll
    for (int h = 0; h < 8; ++h) z[h] = 0.f;
#pragma unroll
    for (int q = 0; q < 32; ++q) acc[q] = 0.f;
    for (int j = tid; j < NN; j += 256) {
        if (!amask[j]) continue;
        const float*  grj = gr + (size_t)j * 8;
        const float4* gj  = (const float4*)(g + (size_t)j * HIDN);
#pragma unroll
        for (int h = 0; h < 8; ++h) {
            float s = gli[h] + grj[h];
            s = (s >= 0.f) ? s : 0.2f * s;
            float w = __expf(s - mh[h]);
            z[h] += w;
            float4 v = gj[h];
            acc[h * 4 + 0] += w * v.x;  acc[h * 4 + 1] += w * v.y;
            acc[h * 4 + 2] += w * v.z;  acc[h * 4 + 3] += w * v.w;
        }
    }
#pragma unroll
    for (int h = 0; h < 8; ++h)  buf[tid * 40 + h] = z[h];
#pragma unroll
    for (int q = 0; q < 32; ++q) buf[tid * 40 + 8 + q] = acc[q];
    __syncthreads();
    for (int s = 128; s > 0; s >>= 1) {
        if (tid < s)
#pragma unroll
            for (int v = 0; v < 40; ++v)
                buf[tid * 40 + v] += buf[(tid + s) * 40 + v];
        __syncthreads();
    }
    if (tid < 32) {
        float o = buf[8 + tid] / buf[tid >> 2];     // head = tid/4
        o = (o > 0.f) ? o : (__expf(o) - 1.f);      // ELU
        if (hf) hf[(size_t)i * HIDN + tid] = o;
        if (hb) hb[(size_t)i * HIDN + tid] = f2bf(o);
    }
}

// ---- layer 3 collapsed path ------------------------------------------------
// w3l = W3 @ a3l, w3r = W3 @ a3r   (W3 is [32][2049] row-major)
__global__ __launch_bounds__(256) void w3_proj_kernel(const float* __restrict__ W3,
                                                      const float* __restrict__ a3l,
                                                      const float* __restrict__ a3r,
                                                      float* __restrict__ w3l,
                                                      float* __restrict__ w3r) {
    __shared__ float bl[256], br[256];
    const int r = blockIdx.x, tid = threadIdx.x;
    float sl = 0.f, sr = 0.f;
    for (int k = tid; k < IND; k += 256) {
        float w = W3[(size_t)r * IND + k];
        sl += w * a3l[k];  sr += w * a3r[k];
    }
    bl[tid] = sl; br[tid] = sr; __syncthreads();
    for (int s = 128; s > 0; s >>= 1) {
        if (tid < s) { bl[tid] += bl[tid + s]; br[tid] += br[tid + s]; }
        __syncthreads();
    }
    if (tid == 0) { w3l[r] = bl[0]; w3r[r] = br[0]; }
}

__global__ void gr3_kernel(const float* __restrict__ h2, const float* __restrict__ w3r,
                           float* __restrict__ gr3) {
    int j = blockIdx.x * 256 + threadIdx.x;
    if (j >= NN) return;
    float s = 0.f;
#pragma unroll
    for (int r = 0; r < HIDN; ++r) s += h2[(size_t)j * HIDN + r] * w3r[r];
    gr3[j] = s;
}

// hbar = (sum_j a_j h2[j,:]) for query row; single block
__global__ __launch_bounds__(256) void l3_agg_kernel(const float* __restrict__ h2,
                                                     const float* __restrict__ w3l,
                                                     const float* __restrict__ gr3,
                                                     const unsigned char* __restrict__ adj,
                                                     const int* __restrict__ curp,
                                                     float* __restrict__ hbar) {
    __shared__ float b3[256 * 33];
    __shared__ unsigned char amask[NN];
    const int tid = threadIdx.x, cur = *curp;

    if (tid < 32) {
        tdm_row_to_lds(adj + (size_t)cur * NN, amask, NN);
        __builtin_amdgcn_s_wait_tensorcnt((short)0);
    }
    __syncthreads();

    float gl3 = 0.f;
#pragma unroll
    for (int r = 0; r < HIDN; ++r) gl3 += h2[(size_t)cur * HIDN + r] * w3l[r];

    float m = -INFINITY;
    for (int j = tid; j < NN; j += 256)
        if (amask[j]) {
            float s = gl3 + gr3[j];  s = (s >= 0.f) ? s : 0.2f * s;
            m = fmaxf(m, s);
        }
    b3[tid] = m; __syncthreads();
    for (int s = 128; s > 0; s >>= 1) {
        if (tid < s) b3[tid] = fmaxf(b3[tid], b3[tid + s]);
        __syncthreads();
    }
    float mh = b3[0]; __syncthreads();

    float z = 0.f, acc[32];
#pragma unroll
    for (int r = 0; r < HIDN; ++r) acc[r] = 0.f;
    for (int j = tid; j < NN; j += 256)
        if (amask[j]) {
            float s = gl3 + gr3[j];  s = (s >= 0.f) ? s : 0.2f * s;
            float w = __expf(s - mh);  z += w;
            const float4* hj = (const float4*)(h2 + (size_t)j * HIDN);
#pragma unroll
            for (int q = 0; q < 8; ++q) {
                float4 v = hj[q];
                acc[q * 4 + 0] += w * v.x;  acc[q * 4 + 1] += w * v.y;
                acc[q * 4 + 2] += w * v.z;  acc[q * 4 + 3] += w * v.w;
            }
        }
    b3[tid * 33] = z;
#pragma unroll
    for (int r = 0; r < HIDN; ++r) b3[tid * 33 + 1 + r] = acc[r];
    __syncthreads();
    for (int s = 128; s > 0; s >>= 1) {
        if (tid < s)
#pragma unroll
            for (int v = 0; v < 33; ++v) b3[tid * 33 + v] += b3[(tid + s) * 33 + v];
        __syncthreads();
    }
    if (tid < HIDN) hbar[tid] = b3[1 + tid] / b3[0];
}

// out3row[k] = hbar . W3[:,k]
__global__ void l3_row_kernel(const float* __restrict__ hbar, const float* __restrict__ W3,
                              float* __restrict__ o3) {
    __shared__ float hb[HIDN];
    int tid = threadIdx.x;
    if (tid < HIDN) hb[tid] = hbar[tid];
    __syncthreads();
    int k = blockIdx.x * 256 + tid;
    if (k < IND) {
        float s = 0.f;
#pragma unroll
        for (int r = 0; r < HIDN; ++r) s += hb[r] * W3[(size_t)r * IND + k];
        o3[k] = s;
    }
}

// hh = leaky(row @ Wf1 + bf1, 0.01); probs = hh @ Wf2 + bf2
__global__ __launch_bounds__(256) void final_fc_kernel(const float* __restrict__ o3,
                                                       const float* __restrict__ rn,
                                                       const float* __restrict__ Wf1,
                                                       const float* __restrict__ bf1,
                                                       const float* __restrict__ Wf2,
                                                       const float* __restrict__ bf2,
                                                       const int* __restrict__ curp,
                                                       float* __restrict__ out) {
    __shared__ float hh[256];
    const int c = threadIdx.x, cur = *curp;
    float s = bf1[c];
    for (int k = 0; k < IND; ++k) s += o3[k] * Wf1[(size_t)k * 256 + c];
    s += rn[cur] * Wf1[(size_t)IND * 256 + c];
    s = (s > 0.f) ? s : 0.01f * s;
    hh[c] = s; __syncthreads();
    if (c < 5) {
        float p = bf2[c];
        for (int q = 0; q < 256; ++q) p += hh[q] * Wf2[(size_t)q * 5 + c];
        out[c] = p;
    }
}

// ---------------------------------------------------------------------------
extern "C" void kernel_launch(void* const* d_in, const int* in_sizes, int n_in,
                              void* d_out, int out_size, void* d_ws, size_t ws_size,
                              hipStream_t stream) {
    const float*         x    = (const float*)d_in[0];
    const unsigned char* adj  = (const unsigned char*)d_in[1];   // jax bool: 1 byte
    const float*         rn   = (const float*)d_in[2];
    const float*         W1   = (const float*)d_in[3];
    const float*         a1l  = (const float*)d_in[4];
    const float*         a1r  = (const float*)d_in[5];
    const float*         W2   = (const float*)d_in[6];
    const float*         a2l  = (const float*)d_in[7];
    const float*         a2r  = (const float*)d_in[8];
    const float*         W3   = (const float*)d_in[9];
    const float*         a3l  = (const float*)d_in[10];
    const float*         a3r  = (const float*)d_in[11];
    const float*         Wf1  = (const float*)d_in[12];
    const float*         bf1  = (const float*)d_in[13];
    const float*         Wf2  = (const float*)d_in[14];
    const float*         bf2  = (const float*)d_in[15];
    const int*           cur  = (const int*)d_in[16];
    float*               out  = (float*)d_out;

    // workspace carve-out (256-B aligned)
    char* ws = (char*)d_ws;
    size_t off = 0;
    auto carve = [&](size_t bytes) -> char* {
        char* p = ws + off;
        off = (off + bytes + 255) & ~(size_t)255;
        return p;
    };
    unsigned short* XB  = (unsigned short*)carve((size_t)NN * KP1 * 2);
    unsigned short* W1T = (unsigned short*)carve((size_t)HIDN * KP1 * 2);
    unsigned short* W2T = (unsigned short*)carve((size_t)HIDN * HIDN * 2);
    unsigned short* H1B = (unsigned short*)carve((size_t)NN * HIDN * 2);
    float* G1   = (float*)carve((size_t)NN * HIDN * 4);
    float* G2   = (float*)carve((size_t)NN * HIDN * 4);
    float* H2   = (float*)carve((size_t)NN * HIDN * 4);
    float* GL1  = (float*)carve((size_t)NN * 8 * 4);
    float* GR1  = (float*)carve((size_t)NN * 8 * 4);
    float* GL2  = (float*)carve((size_t)NN * 8 * 4);
    float* GR2  = (float*)carve((size_t)NN * 8 * 4);
    float* W3L  = (float*)carve(HIDN * 4);
    float* W3R  = (float*)carve(HIDN * 4);
    float* GR3  = (float*)carve(NN * 4);
    float* HBAR = (float*)carve(HIDN * 4);
    float* O3   = (float*)carve(IND * 4);
    if (off > ws_size) return;  // workspace too small: bail deterministically

    // --- stage bf16 operands ---
    pack_x_kernel<<<(NN * KP1 + 255) / 256, 256, 0, stream>>>(x, XB);
    pack_wT_kernel<<<(HIDN * KP1 + 255) / 256, 256, 0, stream>>>(W1, W1T, IND, KP1, HIDN);
    pack_wT_kernel<<<(HIDN * HIDN + 255) / 256, 256, 0, stream>>>(W2, W2T, HIDN, HIDN, HIDN);

    // --- layer 1: g1 = x @ W1 (WMMA), logits, attention+ELU -> h1 (bf16) ---
    gemm_bf16_wmma<<<dim3(NN / 16, HIDN / 16), 32, 0, stream>>>(XB, W1T, G1, KP1, HIDN);
    glr_kernel<<<(NN * 8 + 255) / 256, 256, 0, stream>>>(G1, a1l, a1r, GL1, GR1);
    gat_attn_kernel<<<NN, 256, 0, stream>>>(G1, GL1, GR1, adj, nullptr, H1B);

    // --- layer 2: g2 = h1 @ W2 (WMMA), attention+ELU -> h2 (f32) ---
    gemm_bf16_wmma<<<dim3(NN / 16, HIDN / 16), 32, 0, stream>>>(H1B, W2T, G2, HIDN, HIDN);
    glr_kernel<<<(NN * 8 + 255) / 256, 256, 0, stream>>>(G2, a2l, a2r, GL2, GR2);
    gat_attn_kernel<<<NN, 256, 0, stream>>>(G2, GL2, GR2, adj, H2, nullptr);

    // --- layer 3 (collapsed): only the cur_node row matters ---
    w3_proj_kernel<<<HIDN, 256, 0, stream>>>(W3, a3l, a3r, W3L, W3R);
    gr3_kernel<<<(NN + 255) / 256, 256, 0, stream>>>(H2, W3R, GR3);
    l3_agg_kernel<<<1, 256, 0, stream>>>(H2, W3L, GR3, adj, cur, HBAR);
    l3_row_kernel<<<(IND + 255) / 256, 256, 0, stream>>>(HBAR, W3, O3);

    // --- FC head -> probs[5] ---
    final_fc_kernel<<<1, 256, 0, stream>>>(O3, rn, Wf1, bf1, Wf2, bf2, cur, out);
}
